// GeneralPieceWiseLinearCoupling_36833639530644
// MI455X (gfx1250) — compile-verified, matrix operand order checked by reading, not run on hardware
//
#include <hip/hip_runtime.h>
#include <hip/hip_bf16.h>

// Problem constants (from reference)
#define BATCH    131072
#define PASS     8
#define TRANS    8
#define NBINS    128
#define HIDDEN   256
#define XCOLS    17      // FLOW_SIZE + 1
#define NOUT     1024    // TRANS * NBINS

// Tiling
#define MTILE    64                  // rows per workgroup
#define KC       32                  // K per wmma 16x16x32 step
#define NKC      (HIDDEN / KC)       // 8 K chunks
#define HPAD     264                 // padded f16 row stride for h (528B, 16B-aligned)
#define CPAD     129                 // padded f32 row stride (scan bank spread)

typedef __attribute__((ext_vector_type(16))) _Float16 v16h;
typedef __attribute__((ext_vector_type(8)))  _Float16 v8h;
typedef __attribute__((ext_vector_type(8)))  float    v8f;
typedef __attribute__((ext_vector_type(2)))  float    v2f;

// ---------------------------------------------------------------------------
// Kernel 0: W2 (256 x 1024 f32, k-major) -> W2hT ((t*128+n) x 256 f16, k-contig)
// ---------------------------------------------------------------------------
__global__ __launch_bounds__(256)
void gpwlc_w2t_kernel(const float* __restrict__ W2, _Float16* __restrict__ w2t)
{
    const int col = blockIdx.x;     // 0..1023  (t*128 + n)
    const int k   = threadIdx.x;    // 0..255
    w2t[col * HIDDEN + k] = (_Float16)W2[k * NOUT + col];
}

// ---------------------------------------------------------------------------
// Kernel 1: WMMA-f32 GEMM1 -> WMMA-f16 GEMM2 -> exp/cumsum/gather epilogue
// grid = (BATCH/64, TRANS); one workgroup = 64 rows x one trans-dim (128 bins)
// ---------------------------------------------------------------------------
__global__ __launch_bounds__(256)
void gpwlc_main_kernel(const float* __restrict__ x,
                       const float* __restrict__ W1,
                       const float* __restrict__ b1,
                       const _Float16* __restrict__ w2t,
                       const float* __restrict__ b2,
                       float* __restrict__ out,
                       float* __restrict__ prodbuf)
{
    __shared__ __align__(16) float xAs[MTILE][PASS];                 // 2 KB
    __shared__ __align__(16) float b2s[NBINS];                       // 0.5 KB
    __shared__ __align__(16) float scr[MTILE][4];                    // 1 KB scan combine
    // h tile (f16) and C staging (f32) are never live simultaneously: alias.
    __shared__ __align__(16) unsigned char smemHC[MTILE * HPAD * 2]; // 33.8 KB
    _Float16 (*hAs)[HPAD] = reinterpret_cast<_Float16 (*)[HPAD]>(smemHC);
    float    (*Cs)[CPAD]  = reinterpret_cast<float    (*)[CPAD]>(smemHC);

    const int tid  = threadIdx.x;
    const int wave = tid >> 5;
    const int lane = tid & 31;
    const int t    = blockIdx.y;
    const int row0 = blockIdx.x * MTILE;

    const int lh    = lane & 15;
    const int lhi   = lane >> 4;              // 0 or 1

    // ---- stage xA tile + b2 slice ----
    for (int i = tid; i < MTILE * PASS; i += 256) {
        int r = i >> 3, k = i & 7;
        xAs[r][k] = x[(row0 + r) * XCOLS + k];
    }
    if (tid < NBINS) b2s[tid] = b2[t * NBINS + tid];
    __syncthreads();

    // ================= GEMM1 via V_WMMA_F32_16X16X4_F32 =================
    // wave covers rows (wave>>1)*16..+16, cols (wave&1)*128..+128 of h (64x256)
    {
        const int m0 = (wave >> 1) * 16;
        const int nh = (wave & 1) * 128;
        const int kq = lhi * 2;               // K offset within 4-step: 0 or 2

        // A fragments (xA rows), two K=4 steps covering K=0..7
        v2f af[2];
#pragma unroll
        for (int ks = 0; ks < 2; ++ks) {
            const int koff = ks * 4 + kq;
            af[ks].x = xAs[m0 + lh][koff];
            af[ks].y = xAs[m0 + lh][koff + 1];
        }
        const int mrow = m0 + (lhi ? 8 : 0);
#pragma unroll
        for (int nt = 0; nt < 8; ++nt) {
            const int n = nh + nt * 16 + lh;
            v8f c = (v8f){0.f,0.f,0.f,0.f,0.f,0.f,0.f,0.f};
#pragma unroll
            for (int ks = 0; ks < 2; ++ks) {
                const int koff = ks * 4 + kq;
                v2f bf;
                bf.x = W1[koff * HIDDEN + n];
                bf.y = W1[(koff + 1) * HIDDEN + n];
                c = __builtin_amdgcn_wmma_f32_16x16x4_f32(
                    false, af[ks], false, bf, (short)0, c, false, false);
            }
            const float bias = b1[n];
#pragma unroll
            for (int r = 0; r < 8; ++r)
                hAs[mrow + r][n] = (_Float16)fmaxf(c[r] + bias, 0.0f);
        }
    }
    __syncthreads();

    // ================= GEMM2 via V_WMMA_F32_16X16X32_F16 =================
    const int m_base = (wave >> 1) * 16;      // 0,16,32,48
    const int n_wave = (wave & 1) * 64;       // 0 or 64
    const int khalf  = lhi * 8;               // K split per 16-bit fragment layout

    // load ALL A fragments into registers (8 x v16h = 64 VGPR/lane)
    v16h afrag[NKC];
#pragma unroll
    for (int kc = 0; kc < NKC; ++kc) {
        const _Float16* ap = &hAs[m_base + lh][kc * KC + khalf];
        const v8h alo = *(const v8h*)ap;
        const v8h ahi = *(const v8h*)(ap + 16);
        afrag[kc] = __builtin_shufflevector(alo, ahi,
            0,1,2,3,4,5,6,7,8,9,10,11,12,13,14,15);
    }
    __syncthreads();   // hAs dead after this point -> Cs may reuse the LDS

    v8f acc[4];
#pragma unroll
    for (int i = 0; i < 4; ++i) acc[i] = (v8f){0.f,0.f,0.f,0.f,0.f,0.f,0.f,0.f};

    const _Float16* w2base = w2t + (size_t)(t * NBINS) * HIDDEN;
    __builtin_prefetch(w2base + (size_t)(n_wave + lh) * HIDDEN, 0, 1); // global_prefetch_b8

#pragma unroll
    for (int kc = 0; kc < NKC; ++kc) {
#pragma unroll
        for (int nt = 0; nt < 4; ++nt) {
            const int n = n_wave + nt * 16 + lh;
            const _Float16* bp = w2base + (size_t)n * HIDDEN + kc * KC + khalf;
            const v8h blo = *(const v8h*)bp;
            const v8h bhi = *(const v8h*)(bp + 16);
            const v16h bfrag = __builtin_shufflevector(blo, bhi,
                0,1,2,3,4,5,6,7,8,9,10,11,12,13,14,15);
            acc[nt] = __builtin_amdgcn_wmma_f32_16x16x32_f16(
                false, afrag[kc], false, bfrag, (short)0, acc[nt], false, false);
        }
    }

    // ---- stage C tile to LDS (f32 C layout: vgpr r -> row m_base+8*hi+r) ----
    {
        const int mrow = m_base + (lhi ? 8 : 0);
#pragma unroll
        for (int nt = 0; nt < 4; ++nt) {
            const int n = n_wave + nt * 16 + lh;
#pragma unroll
            for (int r = 0; r < 8; ++r)
                Cs[mrow + r][n] = acc[nt][r];
        }
    }
    __syncthreads();

    // ---- epilogue: exp(logit + b2) in place ----
    for (int i = tid; i < MTILE * NBINS; i += 256) {
        const int m = i >> 7, n = i & 127;
        Cs[m][n] = __expf(Cs[m][n] + b2s[n]);
    }
    __syncthreads();

    // ---- per-row scan, 2 threads/row (64 bins each), combine via LDS ----
    const int m    = tid >> 1;
    const int half = tid & 1;
    float psum = 0.f, pref = 0.f, qb = 0.f, frac = 0.f;
    int bin = 0;
    if (tid < 2 * MTILE) {
        const int rowg = row0 + m;
        const float xb    = x[rowg * XCOLS + PASS + t];
        const float alpha = xb * (float)NBINS;
        const float binf  = floorf(alpha);
        bin = (int)binf;
        bin = bin < 0 ? 0 : (bin > NBINS - 1 ? NBINS - 1 : bin);
        frac = (alpha - binf) * (1.0f / (float)NBINS);

        const int n0 = half * 64;
        for (int n = n0; n < n0 + 64; ++n) {
            const float e = Cs[m][n];
            psum += e;
            if (n < bin)  pref += e;
            if (n == bin) qb = e;
        }
        if (half) { scr[m][0] = psum; scr[m][1] = pref; scr[m][2] = qb; }
    }
    __syncthreads();
    if (tid < 2 * MTILE && half == 0) {
        const int rowg = row0 + m;
        const float total = psum + scr[m][0];
        float prefix, qbv;
        if (bin < 64) { prefix = pref;            qbv = qb; }
        else          { prefix = psum + scr[m][1]; qbv = scr[m][2]; }
        const float inv = 1.0f / total;
        const float cdf_float = qbv * (float)NBINS * inv;   // Q[bin] / (Qnorm/NBINS)
        const float cdf = cdf_float * frac + prefix * inv;  // + Qsum[bin] (leading-0 pad)
        out[rowg * XCOLS + PASS + t] = cdf;
        prodbuf[rowg * TRANS + t]    = cdf_float;
    }
}

// ---------------------------------------------------------------------------
// Kernel 2: xA passthrough + jacobian = x[:,16] * prod_t cdf_float[:,t]
// ---------------------------------------------------------------------------
__global__ __launch_bounds__(256)
void gpwlc_finish_kernel(const float* __restrict__ x,
                         const float* __restrict__ prodbuf,
                         float* __restrict__ out)
{
    const int row = blockIdx.x * 256 + threadIdx.x;
    if (row >= BATCH) return;
    const float* xr   = x + row * XCOLS;
    float*       orow = out + row * XCOLS;
#pragma unroll
    for (int i = 0; i < PASS; ++i) orow[i] = xr[i];
    float p = xr[XCOLS - 1];
#pragma unroll
    for (int t = 0; t < TRANS; ++t) p *= prodbuf[row * TRANS + t];
    orow[XCOLS - 1] = p;
}

extern "C" void kernel_launch(void* const* d_in, const int* in_sizes, int n_in,
                              void* d_out, int out_size, void* d_ws, size_t ws_size,
                              hipStream_t stream) {
    (void)in_sizes; (void)n_in; (void)out_size; (void)ws_size;
    const float* x  = (const float*)d_in[0];
    const float* W1 = (const float*)d_in[1];
    const float* b1 = (const float*)d_in[2];
    const float* W2 = (const float*)d_in[3];
    const float* b2 = (const float*)d_in[4];
    float* out = (float*)d_out;

    // workspace layout: [0, 4MB) prodbuf f32; [4MB, 4.5MB) W2hT f16
    float*    prodbuf = (float*)d_ws;
    _Float16* w2t     = (_Float16*)((char*)d_ws + (size_t)BATCH * TRANS * sizeof(float));

    gpwlc_w2t_kernel<<<NOUT, 256, 0, stream>>>(W2, w2t);

    dim3 grid(BATCH / MTILE, TRANS);
    gpwlc_main_kernel<<<grid, 256, 0, stream>>>(x, W1, b1, w2t, b2, out, prodbuf);

    gpwlc_finish_kernel<<<BATCH / 256, 256, 0, stream>>>(x, prodbuf, out);
}
